// Model_85761906966772
// MI455X (gfx1250) — compile-verified, matrix operand order checked by reading, not run on hardware
//
#include <hip/hip_runtime.h>
#include <hip/hip_bf16.h>

// ---------------------------------------------------------------------------
// Types for CDNA5 WMMA (wave32): v_wmma_f32_16x16x32_f16
// ---------------------------------------------------------------------------
typedef _Float16 v16h __attribute__((ext_vector_type(16)));
typedef float    v8f  __attribute__((ext_vector_type(8)));
typedef _Float16 h2v  __attribute__((ext_vector_type(2)));

__device__ __forceinline__ v8f wmma_f16(v16h a, v16h b, v8f c) {
  return __builtin_amdgcn_wmma_f32_16x16x32_f16(
      /*neg_a=*/false, a, /*neg_b=*/false, b,
      /*c_mod=*/(short)0, c, /*reuse_a=*/false, /*reuse_b=*/false);
}

// 32-bit LDS byte offset of a generic pointer into __shared__ memory
// (AS(3) pointers are the raw LDS offset the async-DMA VDST operand wants).
__device__ __forceinline__ unsigned lds_off(const void* p) {
  return (unsigned)(unsigned long long)(const __attribute__((address_space(3))) char*)p;
}

// CDNA5 async DMA: global memory -> LDS, 16B per lane, tracked by ASYNCcnt.
__device__ __forceinline__ void async_copy_b128(unsigned lds_byte, const void* gptr) {
  asm volatile("global_load_async_to_lds_b128 %0, %1, off"
               :: "v"(lds_byte), "v"(gptr)
               : "memory");
}
__device__ __forceinline__ void wait_async0() {
  asm volatile("s_wait_asynccnt 0x0" ::: "memory");
}

// Load a 16x32 f16 fragment for one lane.
// A layout (16-bit, 16x32): lanes 0-15 row M=lane, K in {0..7,16..23};
// lanes 16-31 row M=lane-16, K in {8..15,24..31}; 2 halves per VGPR.
// B layout assumed mirrored with N <-> M role (row index = N of W[n][k]).
__device__ __forceinline__ v16h load_frag_row(const _Float16* base, int row,
                                              int ld, int kc, int lane) {
  v16h r;
  int hi = (lane >> 4) & 1;
  const _Float16* p = base + (size_t)row * ld + kc + hi * 8;
#pragma unroll
  for (int i = 0; i < 4; ++i) {
    h2v t0 = *(const h2v*)(p + 2 * i);
    r[2 * i]     = t0.x;
    r[2 * i + 1] = t0.y;
    h2v t1 = *(const h2v*)(p + 16 + 2 * i);
    r[8 + 2 * i]     = t1.x;
    r[8 + 2 * i + 1] = t1.y;
  }
  return r;
}

__device__ __forceinline__ float sigf(float x) { return 1.0f / (1.0f + expf(-x)); }

// ---------------------------------------------------------------------------
// Generic converters / packers
// ---------------------------------------------------------------------------

// dst(Npad x Kpad f16) <- src rows n<Nsrc, cols [coloff, coloff+Kcopy) of a
// row-major f32 matrix with row stride Kstride; zero padded elsewhere.
__global__ void k_cvt(_Float16* __restrict__ dst, const float* __restrict__ src,
                      int Nsrc, int Npad, int Kstride, int coloff, int Kcopy, int Kpad) {
  size_t idx = (size_t)blockIdx.x * blockDim.x + threadIdx.x;
  size_t tot = (size_t)Npad * Kpad;
  if (idx >= tot) return;
  int k = (int)(idx % Kpad);
  int n = (int)(idx / Kpad);
  float v = (n < Nsrc && k < Kcopy) ? src[(size_t)n * Kstride + coloff + k] : 0.0f;
  dst[idx] = (_Float16)v;
}

__global__ void k_bias_sum(float* __restrict__ d, const float* __restrict__ a,
                           const float* __restrict__ b, int n) {
  int i = blockIdx.x * blockDim.x + threadIdx.x;
  if (i < n) d[i] = a[i] + b[i];
}

// Pack x (B,T,D) f32 -> rows r = t*B + b, Kpad cols, f16, zero padded.
__global__ void k_pack_bt(_Float16* __restrict__ dst, const float* __restrict__ src,
                          int B_, int T_, int D_, int Kpad) {
  size_t idx = (size_t)blockIdx.x * blockDim.x + threadIdx.x;
  size_t tot = (size_t)T_ * B_ * Kpad;
  if (idx >= tot) return;
  int k = (int)(idx % Kpad);
  size_t r = idx / Kpad;
  int b = (int)(r % B_);
  int t = (int)(r / B_);
  float v = (k < D_) ? src[((size_t)b * T_ + t) * D_ + k] : 0.0f;
  dst[idx] = (_Float16)v;
}

// pyramid + concat(fwd,bwd) + f16 pack (and optional f32 copy).
// hf/hb: (Tin,B,Hp) f32 with Tin = 2*Tout. Output rows r = t'*32+b,
// dim Dout=2*Hp padded to Kpad.
__global__ void k_pyr_pack(const float* __restrict__ hf, const float* __restrict__ hb,
                           _Float16* __restrict__ x16, float* __restrict__ x32,
                           int Tout, int Hp, int Kpad) {
  size_t idx = (size_t)blockIdx.x * blockDim.x + threadIdx.x;
  size_t tot = (size_t)Tout * 32 * Kpad;
  if (idx >= tot) return;
  int k = (int)(idx % Kpad);
  size_t r = idx / Kpad;
  int b  = (int)(r % 32);
  int tp = (int)(r / 32);
  int Dout = 2 * Hp;
  float v = 0.0f;
  if (k < Dout) {
    int t0 = 2 * tp, t1 = 2 * tp + 1;
    if (k < Hp) {
      v = hf[((size_t)t0 * 32 + b) * Hp + k] + hf[((size_t)t1 * 32 + b) * Hp + k];
    } else {
      int kk = k - Hp;
      v = hb[((size_t)t0 * 32 + b) * Hp + kk] + hb[((size_t)t1 * 32 + b) * Hp + kk];
    }
  }
  x16[idx] = (_Float16)v;
  if (x32 && k < Dout) x32[r * (size_t)Dout + k] = v;
}

// ---------------------------------------------------------------------------
// WMMA GEMM: C[M,N] = A[M,K](f16) @ W[N,K](f16)^T + bias0 + bias1 (f32 out)
// M,N multiples of 64; K multiple of 32. Block 256 thr (8 waves), 64x64 tile.
// Tiles staged memory->LDS with CDNA5 async DMA (ASYNCcnt), double buffered
// so the copy of chunk k+1 overlaps the WMMAs of chunk k.
// rowmap=1 remaps output row (t*32+b) -> (b*Tdim+t) (for final logits).
// Nst is both ld and store bound of C.
// ---------------------------------------------------------------------------
__global__ void __launch_bounds__(256) k_gemm(
    const _Float16* __restrict__ A, const _Float16* __restrict__ W,
    const float* __restrict__ bias0, const float* __restrict__ bias1,
    float* __restrict__ C, int M, int N, int K, int Nst, int rowmap, int Tdim) {
  __shared__ _Float16 As[2][64 * 32];
  __shared__ _Float16 Ws[2][64 * 32];
  int m0 = blockIdx.x * 64;
  int n0 = blockIdx.y * 64;
  int tid = threadIdx.x, wave = tid >> 5, lane = tid & 31;
  int msub = wave & 3;
  int nsub2 = (wave >> 2) * 2;
  v8f acc0 = {}, acc1 = {};
  int r = tid >> 2, seg = tid & 3;  // each thread DMAs 16B: row r, cols seg*8..
  const _Float16* gA = A + (size_t)(m0 + r) * K + seg * 8;
  const _Float16* gW = W + (size_t)(n0 + r) * K + seg * 8;
  unsigned la[2] = { lds_off(&As[0][r * 32 + seg * 8]), lds_off(&As[1][r * 32 + seg * 8]) };
  unsigned lw[2] = { lds_off(&Ws[0][r * 32 + seg * 8]), lds_off(&Ws[1][r * 32 + seg * 8]) };
  int nk = K >> 5;
  async_copy_b128(la[0], gA);
  async_copy_b128(lw[0], gW);
  wait_async0();
  __syncthreads();
  for (int ki = 0; ki < nk; ++ki) {
    int cur = ki & 1;
    if (ki + 1 < nk) {
      async_copy_b128(la[cur ^ 1], gA + (size_t)(ki + 1) * 32);
      async_copy_b128(lw[cur ^ 1], gW + (size_t)(ki + 1) * 32);
    }
    const _Float16* Ac = As[cur];
    const _Float16* Wc = Ws[cur];
    v16h a  = load_frag_row(Ac, msub * 16 + (lane & 15), 32, 0, lane);
    v16h b0 = load_frag_row(Wc, nsub2 * 16 + (lane & 15), 32, 0, lane);
    v16h b1 = load_frag_row(Wc, (nsub2 + 1) * 16 + (lane & 15), 32, 0, lane);
    acc0 = wmma_f16(a, b0, acc0);
    acc1 = wmma_f16(a, b1, acc1);
    wait_async0();
    __syncthreads();
  }
#pragma unroll
  for (int v = 0; v < 8; ++v) {
    int mm = m0 + msub * 16 + v + ((lane >> 4) << 3);
    int mmO = rowmap ? ((mm & 31) * Tdim + (mm >> 5)) : mm;
    int nnA = n0 + nsub2 * 16 + (lane & 15);
    int nnB = nnA + 16;
    if (nnA < Nst) {
      float bs = (bias0 ? bias0[nnA] : 0.0f) + (bias1 ? bias1[nnA] : 0.0f);
      C[(size_t)mmO * Nst + nnA] = acc0[v] + bs;
    }
    if (nnB < Nst) {
      float bs = (bias0 ? bias0[nnB] : 0.0f) + (bias1 ? bias1[nnB] : 0.0f);
      C[(size_t)mmO * Nst + nnB] = acc1[v] + bs;
    }
  }
}

// ---------------------------------------------------------------------------
// Persistent LSTM recurrence (one workgroup per direction).
// gin: (T, B=32, 4*Hp) f32, biases already folded in.
// whh: (4*Hp, Hp) f16.  hout: (T, 32, Hp) f32.
// h kept in LDS as f16; c kept in registers in WMMA C-fragment layout.
// Block = 512 threads (16 waves). Hp in {256,512}.
// ---------------------------------------------------------------------------
__global__ void __launch_bounds__(512) k_lstm_rec(
    const float* __restrict__ gin0, const float* __restrict__ gin1,
    const _Float16* __restrict__ whh0, const _Float16* __restrict__ whh1,
    float* __restrict__ hout0, float* __restrict__ hout1, int T, int Hp) {
  extern __shared__ char smem[];
  _Float16* h16 = (_Float16*)smem;  // 32*Hp
  int dir = blockIdx.x;
  const float* gin = dir ? gin1 : gin0;
  const _Float16* whh = dir ? whh1 : whh0;
  float* hout = dir ? hout1 : hout0;
  int tid = threadIdx.x, wave = tid >> 5, lane = tid & 31;
  int pairs = (Hp >> 4) * 2;  // 2 M-tiles x (Hp/16) N0-tiles
  int G4 = 4 * Hp;
  v8f c_st[4] = {};
  float hreg[4][8];
  for (int i = tid; i < 32 * Hp; i += blockDim.x) h16[i] = (_Float16)0.0f;
  __syncthreads();
  for (int step = 0; step < T; ++step) {
    int t = dir ? (T - 1 - step) : step;
    // prefetch next step's gate inputs (gin streamed once from HBM)
    if (step + 1 < T) {
      int tn = dir ? (T - 2 - step) : (step + 1);
      const float* pf = gin + ((size_t)tn * 32) * G4;
      __builtin_prefetch(pf + (size_t)tid * 32, 0, 3);
    }
    int pi = 0;
    for (int p = wave; p < pairs; p += 16, ++pi) {
      int mt = p & 1, n0t = p >> 1;
      v8f acc[4] = {};
#pragma unroll
      for (int g = 0; g < 4; ++g) {
        int nrow = g * Hp + n0t * 16 + (lane & 15);
        for (int kc = 0; kc < Hp; kc += 32) {
          v16h a = load_frag_row(h16, mt * 16 + (lane & 15), Hp, kc, lane);
          v16h b = load_frag_row(whh, nrow, Hp, kc, lane);
          acc[g] = wmma_f16(a, b, acc[g]);
        }
      }
#pragma unroll
      for (int v = 0; v < 8; ++v) {
        int b_ = mt * 16 + v + ((lane >> 4) << 3);
        int n_ = n0t * 16 + (lane & 15);
        size_t base = ((size_t)t * 32 + b_) * G4;
        float gi = acc[0][v] + gin[base + n_];
        float gf = acc[1][v] + gin[base + Hp + n_];
        float gg = acc[2][v] + gin[base + 2 * Hp + n_];
        float go = acc[3][v] + gin[base + 3 * Hp + n_];
        float c = sigf(gf) * c_st[pi][v] + sigf(gi) * tanhf(gg);
        c_st[pi][v] = c;
        hreg[pi][v] = sigf(go) * tanhf(c);
      }
    }
    __syncthreads();  // all waves finished reading h16 for this step
    pi = 0;
    for (int p = wave; p < pairs; p += 16, ++pi) {
      int mt = p & 1, n0t = p >> 1;
#pragma unroll
      for (int v = 0; v < 8; ++v) {
        int b_ = mt * 16 + v + ((lane >> 4) << 3);
        int n_ = n0t * 16 + (lane & 15);
        h16[b_ * Hp + n_] = (_Float16)hreg[pi][v];
        hout[((size_t)t * 32 + b_) * Hp + n_] = hreg[pi][v];
      }
    }
    __syncthreads();
  }
}

// ---------------------------------------------------------------------------
// Persistent attention decoder (d1). One workgroup, 512 threads (16 waves).
// Per step: thW = h@W1h^T (WMMA) -> e = tanh(W1x_pre + thW)@W2 + b2 ->
// softmax over S=128 -> ctx -> gates = ctx@Wihc^T + h@Whh^T (+one-hot col,
// +bias) -> cell update. Writes a1 as f16, rows t*32+b.
// ---------------------------------------------------------------------------
__global__ void __launch_bounds__(512) k_decoder(
    const float* __restrict__ w1xpre,      // (128*32, 512), b1 folded in
    const _Float16* __restrict__ w1h16,    // (512,512)
    const float* __restrict__ w2, const float* __restrict__ b2,
    const float* __restrict__ xe32,        // (128*32, 512)
    const _Float16* __restrict__ wihc16,   // (2048,512)
    const _Float16* __restrict__ whhd16,   // (2048,512)
    const float* __restrict__ wih_d1_full, // (2048,1512) f32
    const float* __restrict__ bsum_d1,     // 2048
    const int* __restrict__ y,             // (32,128)
    _Float16* __restrict__ a1_16) {        // (128*32,512)
  extern __shared__ char smem[];
  _Float16* h16   = (_Float16*)smem;     // 32*512
  _Float16* thW   = h16 + 32 * 512;      // 32*512
  _Float16* ctx16 = thW + 32 * 512;      // 32*512
  float* a_s = (float*)(ctx16 + 32 * 512);  // 32*128
  float* w2s = a_s + 32 * 128;              // 512
  int tid = threadIdx.x, wave = tid >> 5, lane = tid & 31;
  for (int i = tid; i < 32 * 512; i += blockDim.x) h16[i] = (_Float16)0.0f;
  for (int i = tid; i < 512; i += blockDim.x) w2s[i] = w2[i];
  float b2v = b2[0];
  v8f c_st[4] = {};
  float hreg[4][8];
  __syncthreads();
  for (int t = 0; t < 128; ++t) {
    // (a) thW[b][j] = (h @ W1h^T)[b][j]
    for (int p = wave; p < 64; p += 16) {
      int mt = p & 1, jt = p >> 1;
      v8f acc = {};
      for (int kc = 0; kc < 512; kc += 32) {
        v16h a = load_frag_row(h16, mt * 16 + (lane & 15), 512, kc, lane);
        v16h b = load_frag_row(w1h16, jt * 16 + (lane & 15), 512, kc, lane);
        acc = wmma_f16(a, b, acc);
      }
#pragma unroll
      for (int v = 0; v < 8; ++v) {
        int b_ = mt * 16 + v + ((lane >> 4) << 3);
        int j_ = jt * 16 + (lane & 15);
        thW[b_ * 512 + j_] = (_Float16)acc[v];
      }
    }
    __syncthreads();
    // (b) energies + softmax over s (wave handles batch rows)
    for (int b_ = wave; b_ < 32; b_ += 16) {
      float ev[4];
#pragma unroll
      for (int q = 0; q < 4; ++q) {
        int s = lane + 32 * q;
        const float* wx = w1xpre + ((size_t)s * 32 + b_) * 512;
        const _Float16* th = thW + b_ * 512;
        float acc = 0.0f;
        for (int j = 0; j < 512; ++j) acc += w2s[j] * tanhf(wx[j] + (float)th[j]);
        ev[q] = acc + b2v;
      }
      float m = fmaxf(fmaxf(ev[0], ev[1]), fmaxf(ev[2], ev[3]));
      for (int off = 16; off >= 1; off >>= 1) m = fmaxf(m, __shfl_xor(m, off, 32));
      float ex[4], ss = 0.0f;
#pragma unroll
      for (int q = 0; q < 4; ++q) { ex[q] = expf(ev[q] - m); ss += ex[q]; }
      for (int off = 16; off >= 1; off >>= 1) ss += __shfl_xor(ss, off, 32);
#pragma unroll
      for (int q = 0; q < 4; ++q) a_s[b_ * 128 + lane + 32 * q] = ex[q] / ss;
    }
    __syncthreads();
    // (c) ctx[b][j] = sum_s a[b][s] * xe[s][b][j]
    for (int b_ = wave; b_ < 32; b_ += 16) {
      float acc[16];
#pragma unroll
      for (int jj = 0; jj < 16; ++jj) acc[jj] = 0.0f;
      for (int s = 0; s < 128; ++s) {
        float as = a_s[b_ * 128 + s];
        const float* xr = xe32 + ((size_t)s * 32 + b_) * 512;
#pragma unroll
        for (int jj = 0; jj < 16; ++jj) acc[jj] += as * xr[lane + 32 * jj];
      }
#pragma unroll
      for (int jj = 0; jj < 16; ++jj)
        ctx16[b_ * 512 + lane + 32 * jj] = (_Float16)acc[jj];
    }
    __syncthreads();
    // (d) d1 gates: ctx@Wihc^T + h@Whh^T (+one-hot col + bias), cell update
    int pi = 0;
    for (int p = wave; p < 64; p += 16, ++pi) {
      int mt = p & 1, n0t = p >> 1;
      v8f acc[4] = {};
#pragma unroll
      for (int g = 0; g < 4; ++g) {
        int nrow = g * 512 + n0t * 16 + (lane & 15);
        for (int kc = 0; kc < 512; kc += 32) {
          v16h a = load_frag_row(ctx16, mt * 16 + (lane & 15), 512, kc, lane);
          v16h b = load_frag_row(wihc16, nrow, 512, kc, lane);
          acc[g] = wmma_f16(a, b, acc[g]);
        }
        for (int kc = 0; kc < 512; kc += 32) {
          v16h a = load_frag_row(h16, mt * 16 + (lane & 15), 512, kc, lane);
          v16h b = load_frag_row(whhd16, nrow, 512, kc, lane);
          acc[g] = wmma_f16(a, b, acc[g]);
        }
      }
#pragma unroll
      for (int v = 0; v < 8; ++v) {
        int b_ = mt * 16 + v + ((lane >> 4) << 3);
        int n_ = n0t * 16 + (lane & 15);
        int tok = (t == 0) ? 0 : y[b_ * 128 + (t - 1)];
        float gv[4];
#pragma unroll
        for (int g = 0; g < 4; ++g) {
          int nr = g * 512 + n_;
          gv[g] = acc[g][v] + wih_d1_full[(size_t)nr * 1512 + 512 + tok] + bsum_d1[nr];
        }
        float c = sigf(gv[1]) * c_st[pi][v] + sigf(gv[0]) * tanhf(gv[2]);
        c_st[pi][v] = c;
        hreg[pi][v] = sigf(gv[3]) * tanhf(c);
      }
    }
    __syncthreads();
    pi = 0;
    for (int p = wave; p < 64; p += 16, ++pi) {
      int mt = p & 1, n0t = p >> 1;
#pragma unroll
      for (int v = 0; v < 8; ++v) {
        int b_ = mt * 16 + v + ((lane >> 4) << 3);
        int n_ = n0t * 16 + (lane & 15);
        float h = hreg[pi][v];
        h16[b_ * 512 + n_] = (_Float16)h;
        a1_16[((size_t)t * 32 + b_) * 512 + n_] = (_Float16)h;
      }
    }
    __syncthreads();
  }
}

// ---------------------------------------------------------------------------
// Loss: per-row softmax CE, then deterministic fixed-order final reduction.
// ---------------------------------------------------------------------------
__global__ void k_loss_row(const float* __restrict__ out, const int* __restrict__ y,
                           const float* __restrict__ pad, float* __restrict__ lrow) {
  int g = blockIdx.x;  // g = b*128 + t
  const float* row = out + (size_t)g * 1000;
  __shared__ float red[256];
  int tid = threadIdx.x;
  float m = -1e30f;
  for (int i = tid; i < 1000; i += 256) m = fmaxf(m, row[i]);
  red[tid] = m; __syncthreads();
  for (int s = 128; s > 0; s >>= 1) {
    if (tid < s) red[tid] = fmaxf(red[tid], red[tid + s]);
    __syncthreads();
  }
  m = red[0]; __syncthreads();
  float sum = 0.0f;
  for (int i = tid; i < 1000; i += 256) sum += expf(row[i] - m);
  red[tid] = sum; __syncthreads();
  for (int s = 128; s > 0; s >>= 1) {
    if (tid < s) red[tid] += red[tid + s];
    __syncthreads();
  }
  if (tid == 0) {
    float p = expf(row[y[g]] - m) / red[0];
    lrow[g] = -logf(p + 1e-8f) * pad[g];
  }
}

__global__ void k_loss_final(const float* __restrict__ lrow,
                             const float* __restrict__ pad, float* __restrict__ ol) {
  __shared__ float r1[1024], r2[1024];
  int tid = threadIdx.x;
  float a = 0.0f, b = 0.0f;
  for (int i = tid; i < 4096; i += 1024) { a += lrow[i]; b += pad[i]; }
  r1[tid] = a; r2[tid] = b; __syncthreads();
  for (int s = 512; s > 0; s >>= 1) {
    if (tid < s) { r1[tid] += r1[tid + s]; r2[tid] += r2[tid + s]; }
    __syncthreads();
  }
  if (tid == 0) ol[0] = r1[0] / (r2[0] + 1e-8f);
}

// ---------------------------------------------------------------------------
// Host orchestration
// ---------------------------------------------------------------------------
extern "C" void kernel_launch(void* const* d_in, const int* in_sizes, int n_in,
                              void* d_out, int out_size, void* d_ws, size_t ws_size,
                              hipStream_t stream) {
  (void)in_sizes; (void)n_in; (void)out_size; (void)ws_size;
  const float* x   = (const float*)d_in[0];   // (32,1024,80)
  const int*   y   = (const int*)d_in[1];     // (32,128)
  const float* pad = (const float*)d_in[2];   // (32,128)
  // enc1.f=3..6 enc1.b=7..10 enc2.f=11..14 enc2.b=15..18 enc3.f=19..22
  // enc3.b=23..26 d1=27..30 d2=31..34 att.W1=35 b1=36 W2=37 b2=38 lin.W=39 b=40
  auto F = [&](int i) { return (const float*)d_in[i]; };

  char* wsp = (char*)d_ws;
  size_t off = 0;
  auto alloc = [&](size_t bytes) -> char* {
    char* p = wsp + off;
    off += (bytes + 255) & ~(size_t)255;
    return p;
  };
  typedef _Float16 h16t;
  // f16 weights
  h16t* e1f_wih = (h16t*)alloc(1024ull *  96 * 2);
  h16t* e1f_whh = (h16t*)alloc(1024ull * 256 * 2);
  h16t* e1b_wih = (h16t*)alloc(1024ull *  96 * 2);
  h16t* e1b_whh = (h16t*)alloc(1024ull * 256 * 2);
  h16t* e2f_wih = (h16t*)alloc(2048ull * 512 * 2);
  h16t* e2f_whh = (h16t*)alloc(2048ull * 512 * 2);
  h16t* e2b_wih = (h16t*)alloc(2048ull * 512 * 2);
  h16t* e2b_whh = (h16t*)alloc(2048ull * 512 * 2);
  h16t* e3f_wih = (h16t*)alloc(1024ull * 1024 * 2);
  h16t* e3f_whh = (h16t*)alloc(1024ull *  256 * 2);
  h16t* e3b_wih = (h16t*)alloc(1024ull * 1024 * 2);
  h16t* e3b_whh = (h16t*)alloc(1024ull *  256 * 2);
  h16t* d1_wihc = (h16t*)alloc(2048ull * 512 * 2);
  h16t* d1_whh  = (h16t*)alloc(2048ull * 512 * 2);
  h16t* d2_wih  = (h16t*)alloc(2048ull * 512 * 2);
  h16t* d2_whh  = (h16t*)alloc(2048ull * 512 * 2);
  h16t* w1x16   = (h16t*)alloc(512ull * 512 * 2);
  h16t* w1h16   = (h16t*)alloc(512ull * 512 * 2);
  h16t* lin16   = (h16t*)alloc(1024ull * 512 * 2);
  // bias sums
  float* bs_e1f = (float*)alloc(1024 * 4);
  float* bs_e1b = (float*)alloc(1024 * 4);
  float* bs_e2f = (float*)alloc(2048 * 4);
  float* bs_e2b = (float*)alloc(2048 * 4);
  float* bs_e3f = (float*)alloc(1024 * 4);
  float* bs_e3b = (float*)alloc(1024 * 4);
  float* bs_d1  = (float*)alloc(2048 * 4);
  float* bs_d2  = (float*)alloc(2048 * 4);
  // activations
  h16t*  x16A   = (h16t*)alloc(8192ull * 1024 * 2);   // enc1 in / enc3 in
  h16t*  x16B   = (h16t*)alloc(16384ull * 512 * 2);   // enc2 in / xe16
  float* ginF   = (float*)alloc(32768ull * 1024 * 4); // also enc2 (16384x2048), d2
  float* ginB   = (float*)alloc(32768ull * 1024 * 4);
  float* hF     = (float*)alloc(32768ull * 256 * 4);  // also enc2 (16384x512), a2
  float* hB     = (float*)alloc(32768ull * 256 * 4);
  float* xe32   = (float*)alloc(4096ull * 512 * 4);
  float* w1xpre = (float*)alloc(4096ull * 512 * 4);
  h16t*  a1_16  = (h16t*)alloc(4096ull * 512 * 2);
  h16t*  a2_16  = (h16t*)alloc(4096ull * 512 * 2);
  float* lrow   = (float*)alloc(4096 * 4);

  auto cvt = [&](h16t* dst, const float* src, int Ns, int Np, int Kst, int co,
                 int Kc, int Kp) {
    size_t tot = (size_t)Np * Kp;
    k_cvt<<<(unsigned)((tot + 255) / 256), 256, 0, stream>>>(dst, src, Ns, Np, Kst, co, Kc, Kp);
  };
  auto bsum = [&](float* d, const float* a, const float* b, int n) {
    k_bias_sum<<<(n + 255) / 256, 256, 0, stream>>>(d, a, b, n);
  };
  auto gemm = [&](const h16t* A, const h16t* W, const float* b0, const float* b1,
                  float* C, int M, int N, int K, int Nst, int rowmap, int Td) {
    dim3 g(M / 64, N / 64);
    k_gemm<<<g, 256, 0, stream>>>(A, W, b0, b1, C, M, N, K, Nst, rowmap, Td);
  };

  // --- weight conversion ---
  cvt(e1f_wih, F(3), 1024, 1024, 80, 0, 80, 96);
  cvt(e1f_whh, F(4), 1024, 1024, 256, 0, 256, 256);
  cvt(e1b_wih, F(7), 1024, 1024, 80, 0, 80, 96);
  cvt(e1b_whh, F(8), 1024, 1024, 256, 0, 256, 256);
  cvt(e2f_wih, F(11), 2048, 2048, 512, 0, 512, 512);
  cvt(e2f_whh, F(12), 2048, 2048, 512, 0, 512, 512);
  cvt(e2b_wih, F(15), 2048, 2048, 512, 0, 512, 512);
  cvt(e2b_whh, F(16), 2048, 2048, 512, 0, 512, 512);
  cvt(e3f_wih, F(19), 1024, 1024, 1024, 0, 1024, 1024);
  cvt(e3f_whh, F(20), 1024, 1024, 256, 0, 256, 256);
  cvt(e3b_wih, F(23), 1024, 1024, 1024, 0, 1024, 1024);
  cvt(e3b_whh, F(24), 1024, 1024, 256, 0, 256, 256);
  cvt(d1_wihc, F(27), 2048, 2048, 1512, 0, 512, 512);
  cvt(d1_whh,  F(28), 2048, 2048, 512, 0, 512, 512);
  cvt(d2_wih,  F(31), 2048, 2048, 512, 0, 512, 512);
  cvt(d2_whh,  F(32), 2048, 2048, 512, 0, 512, 512);
  cvt(w1x16, F(35), 512, 512, 1024, 0, 512, 512);
  cvt(w1h16, F(35), 512, 512, 1024, 512, 512, 512);
  cvt(lin16, F(39), 1000, 1024, 512, 0, 512, 512);
  bsum(bs_e1f, F(5), F(6), 1024);
  bsum(bs_e1b, F(9), F(10), 1024);
  bsum(bs_e2f, F(13), F(14), 2048);
  bsum(bs_e2b, F(17), F(18), 2048);
  bsum(bs_e3f, F(21), F(22), 1024);
  bsum(bs_e3b, F(25), F(26), 1024);
  bsum(bs_d1, F(29), F(30), 2048);
  bsum(bs_d2, F(33), F(34), 2048);

  // --- enc1: T=1024, Din=80->96, Hp=256 ---
  {
    size_t tot = 32768ull * 96;
    k_pack_bt<<<(unsigned)((tot + 255) / 256), 256, 0, stream>>>(x16A, x, 32, 1024, 80, 96);
  }
  gemm(x16A, e1f_wih, bs_e1f, nullptr, ginF, 32768, 1024, 96, 1024, 0, 0);
  gemm(x16A, e1b_wih, bs_e1b, nullptr, ginB, 32768, 1024, 96, 1024, 0, 0);
  k_lstm_rec<<<2, 512, 32 * 256 * 2, stream>>>(ginF, ginB, e1f_whh, e1b_whh, hF, hB, 1024, 256);
  {
    size_t tot = 512ull * 32 * 512;
    k_pyr_pack<<<(unsigned)((tot + 255) / 256), 256, 0, stream>>>(hF, hB, x16B, nullptr, 512, 256, 512);
  }
  // --- enc2: T=512, Din=512, Hp=512 ---
  gemm(x16B, e2f_wih, bs_e2f, nullptr, ginF, 16384, 2048, 512, 2048, 0, 0);
  gemm(x16B, e2b_wih, bs_e2b, nullptr, ginB, 16384, 2048, 512, 2048, 0, 0);
  k_lstm_rec<<<2, 512, 32 * 512 * 2, stream>>>(ginF, ginB, e2f_whh, e2b_whh, hF, hB, 512, 512);
  {
    size_t tot = 256ull * 32 * 1024;
    k_pyr_pack<<<(unsigned)((tot + 255) / 256), 256, 0, stream>>>(hF, hB, x16A, nullptr, 256, 512, 1024);
  }
  // --- enc3: T=256, Din=1024, Hp=256 ---
  gemm(x16A, e3f_wih, bs_e3f, nullptr, ginF, 8192, 1024, 1024, 1024, 0, 0);
  gemm(x16A, e3b_wih, bs_e3b, nullptr, ginB, 8192, 1024, 1024, 1024, 0, 0);
  k_lstm_rec<<<2, 512, 32 * 256 * 2, stream>>>(ginF, ginB, e3f_whh, e3b_whh, hF, hB, 256, 256);
  {
    size_t tot = 128ull * 32 * 512;  // xe16 (x16B) + xe32
    k_pyr_pack<<<(unsigned)((tot + 255) / 256), 256, 0, stream>>>(hF, hB, x16B, xe32, 128, 256, 512);
  }
  // --- attention precompute: W1x_pre = xe @ W1x^T + b1 ---
  gemm(x16B, w1x16, F(36), nullptr, w1xpre, 4096, 512, 512, 512, 0, 0);
  // --- decoder (d1 + attention), persistent single workgroup ---
  {
    size_t sh = (size_t)(3 * 32 * 512) * 2 + (size_t)(32 * 128 + 512) * 4;  // 116736 B
    k_decoder<<<1, 512, sh, stream>>>(w1xpre, w1h16, F(37), F(38), xe32,
                                      d1_wihc, d1_whh, F(27), bs_d1, y, a1_16);
  }
  // --- d2 LSTM over a1 ---
  gemm(a1_16, d2_wih, bs_d2, nullptr, ginF, 4096, 2048, 512, 2048, 0, 0);
  k_lstm_rec<<<1, 512, 32 * 512 * 2, stream>>>(ginF, ginF, d2_whh, d2_whh, hF, hF, 128, 512);
  cvt(a2_16, hF, 4096, 4096, 512, 0, 512, 512);
  // --- output projection: out[b,t,:] (rowmap t*32+b -> b*128+t) ---
  gemm(a2_16, lin16, F(40), nullptr, (float*)d_out, 4096, 1024, 512, 1000, 1, 128);
  // --- loss ---
  k_loss_row<<<4096, 256, 0, stream>>>((const float*)d_out, y, pad, lrow);
  k_loss_final<<<1, 1024, 0, stream>>>(lrow, pad, (float*)d_out + 4096000);
}
